// FeatureFusionModule_62560493633650
// MI455X (gfx1250) — matrix operand order repaired and checked by reading, not compiled, not run on hardware
//
#include <hip/hip_runtime.h>

typedef __bf16 bf16_t;
typedef __attribute__((ext_vector_type(16))) __bf16 bf16x16;
typedef __attribute__((ext_vector_type(8)))  float  floatx8;

constexpr int kB = 64, kC = 512, kH = 20, kW = 15, kN = 300;
constexpr int kRows = kB * kN;          // 19200
constexpr long long kBCN = (long long)kB * kC * kN; // 9,830,400

// ---------------------------------------------------------------------------
// WMMA helpers (gfx1250, wave32). Fragment layouts per CDNA5 ISA 7.12.2.
// ---------------------------------------------------------------------------
__device__ inline floatx8 wmma_bf16(bf16x16 a, bf16x16 b, floatx8 c) {
  return __builtin_amdgcn_wmma_f32_16x16x32_bf16(false, a, false, b, (short)0, c,
                                                 false, false);
}

union Frag16 {
  bf16x16 v;
  unsigned int u[8];
};

// A fragment 16(M)x32(K) from LDS tile stored [M][K], row stride (elements)
// must be even.  lane<16: row=lane, K {0..7,16..23}; lane>=16: row=lane-16,
// K {8..15,24..31}.  Vector element pair (2j,2j+1) == VGPR j == one u32 load.
__device__ inline bf16x16 load_a_frag(int lane, const bf16_t* tile, int stride) {
  const int row = lane & 15;
  const int kb2 = (lane & 16) ? 4 : 0;  // K-base / 2
  const unsigned int* p = (const unsigned int*)(tile + row * stride);
  Frag16 f;
#pragma unroll
  for (int j = 0; j < 4; ++j) {
    f.u[j]     = p[kb2 + j];       // K = kb + 2j, kb + 2j + 1
    f.u[4 + j] = p[8 + kb2 + j];   // K = 16 + kb + 2j, +1
  }
  return f.v;
}

// B fragment 32(K)x16(N) from LDS tile stored [N][K], row stride even.
// lane<16: K {0..7,16..23}; lane>=16: K {8..15,24..31}; col = lane&15.
__device__ inline bf16x16 load_b_frag(int lane, const bf16_t* tile, int stride) {
  const int col = lane & 15;
  const int kb2 = (lane & 16) ? 4 : 0;
  const unsigned int* p = (const unsigned int*)(tile + col * stride);
  Frag16 f;
#pragma unroll
  for (int j = 0; j < 4; ++j) {
    f.u[j]     = p[kb2 + j];
    f.u[4 + j] = p[8 + kb2 + j];
  }
  return f.v;
}

// C/D fragment: VGPR r -> M = r + 8*(lane>=16), N = lane&15.
template <typename ST>
__device__ inline void store_c(int lane, const floatx8& c, ST st) {
  const int col = lane & 15;
  const int rb  = (lane & 16) ? 8 : 0;
#pragma unroll
  for (int r = 0; r < 8; ++r) st(rb + r, col, c[r]);
}

// ---------------------------------------------------------------------------
// Generic tiled GEMM: D(M x Nc) = A(M x K, row-major) * B(Nc x K, row-major)^T
// 64x64 block tile, 8 waves (4 row x 2 col), each wave 16x32 (2 WMMA tiles).
// Requires K % 4 == 0, lda/ldb % 4 == 0 elements (8B-aligned vector fills).
// ---------------------------------------------------------------------------
template <int OUT_BF16>
__global__ __launch_bounds__(256) void gemm_kernel(
    const bf16_t* __restrict__ A, int lda, long long sA,
    const bf16_t* __restrict__ Bw, int ldb, long long sB,
    void* __restrict__ D, int ldd, long long sD,
    const float* __restrict__ bias_row, const float* __restrict__ bias_col,
    int M, int Nc, int K) {
  __shared__ bf16_t at[64][36];
  __shared__ bf16_t bt[64][36];
  const int bz = blockIdx.z;
  A  += (size_t)bz * sA;
  Bw += (size_t)bz * sB;
  const size_t dOff = (size_t)bz * sD;
  const int row0 = blockIdx.x * 64, col0 = blockIdx.y * 64;
  const int tid = threadIdx.x, wave = tid >> 5, lane = tid & 31;
  const int wrow = wave & 3, wcol = wave >> 2;
  floatx8 acc0 = {}, acc1 = {};
  for (int k0 = 0; k0 < K; k0 += 32) {
    for (int i = tid; i < 64 * 8; i += 256) {
      const int r = i >> 3, kk = (i & 7) << 2;
      const int gk = k0 + kk;
      unsigned long long va = 0ull, vb = 0ull;
      const int gr = row0 + r;
      if (gr < M && gk < K)
        va = *(const unsigned long long*)(A + (size_t)gr * lda + gk);
      const int gc = col0 + r;
      if (gc < Nc && gk < K)
        vb = *(const unsigned long long*)(Bw + (size_t)gc * ldb + gk);
      *(unsigned long long*)&at[r][kk] = va;
      *(unsigned long long*)&bt[r][kk] = vb;
    }
    __syncthreads();
    bf16x16 a  = load_a_frag(lane, &at[wrow * 16][0], 36);
    bf16x16 b0 = load_b_frag(lane, &bt[wcol * 32][0], 36);
    bf16x16 b1 = load_b_frag(lane, &bt[wcol * 32 + 16][0], 36);
    acc0 = wmma_bf16(a, b0, acc0);
    acc1 = wmma_bf16(a, b1, acc1);
    __syncthreads();
  }
  const int rowb = row0 + wrow * 16;
  const int colb = col0 + wcol * 32;
  auto emit = [&](const floatx8& c, int coff) {
    store_c(lane, c, [&](int r, int cc, float v) {
      const int gr = rowb + r, gc = colb + coff + cc;
      if (gr < M && gc < Nc) {
        float o = v;
        if (bias_row) o += bias_row[gr];
        if (bias_col) o += bias_col[gc];
        if (OUT_BF16)
          ((bf16_t*)D)[dOff + (size_t)gr * ldd + gc] = (bf16_t)o;
        else
          ((float*)D)[dOff + (size_t)gr * ldd + gc] = o;
      }
    });
  };
  emit(acc0, 0);
  emit(acc1, 16);
}

// ---------------------------------------------------------------------------
// Pack x (B,C,H,W) f32 -> T (B*N, C) row-major bf16
// ---------------------------------------------------------------------------
__global__ __launch_bounds__(256) void pack_kernel(const float* __restrict__ x,
                                                   bf16_t* __restrict__ T) {
  const size_t i = (size_t)blockIdx.x * 256 + threadIdx.x;
  if (i >= (size_t)kBCN) return;
  const int ch = (int)(i % kC);
  const size_t t = i / kC;
  const int n = (int)(t % kN);
  const int b = (int)(t / kN);
  T[i] = (bf16_t)x[((size_t)b * kC + ch) * kN + n];
}

__global__ __launch_bounds__(256) void f32_to_bf16_kernel(const float* __restrict__ s,
                                                          bf16_t* __restrict__ d, int n) {
  const int i = blockIdx.x * 256 + threadIdx.x;
  if (i < n) d[i] = (bf16_t)s[i];
}

// ---------------------------------------------------------------------------
// Row LayerNorm(1024) + ReLU + split: cols 0..511 -> Cc (B,C,N) channel-major,
// cols 512..1023 -> S (B*N, 512) row-major.  One block per row.
// ---------------------------------------------------------------------------
__global__ __launch_bounds__(256) void ln_relu_split_kernel(
    const bf16_t* __restrict__ Y, const float* __restrict__ lnw,
    const float* __restrict__ lnb, bf16_t* __restrict__ S, bf16_t* __restrict__ Cc) {
  __shared__ float r1[256], r2[256];
  const int g = blockIdx.x, tid = threadIdx.x;
  const int b = g / kN, n = g % kN;
  const bf16_t* row = Y + (size_t)g * 1024;
  float v[4], s = 0.f, s2 = 0.f;
#pragma unroll
  for (int j = 0; j < 4; ++j) {
    const float x = (float)row[tid + j * 256];
    v[j] = x; s += x; s2 += x * x;
  }
  r1[tid] = s; r2[tid] = s2;
  __syncthreads();
  for (int off = 128; off; off >>= 1) {
    if (tid < off) { r1[tid] += r1[tid + off]; r2[tid] += r2[tid + off]; }
    __syncthreads();
  }
  const float mean = r1[0] * (1.f / 1024.f);
  const float var  = r2[0] * (1.f / 1024.f) - mean * mean;
  const float inv  = rsqrtf(var + 1e-5f);
#pragma unroll
  for (int j = 0; j < 4; ++j) {
    const int c = tid + j * 256;
    float o = (v[j] - mean) * inv * lnw[c] + lnb[c];
    o = fmaxf(o, 0.f);
    if (c < 512)
      Cc[((size_t)b * kC + c) * kN + n] = (bf16_t)o;
    else
      S[(size_t)g * kC + (c - 512)] = (bf16_t)o;
  }
}

// ---------------------------------------------------------------------------
// ctx = softmax(K^T V * scale, axis=rows).  One workgroup per (b,h).
// K/V loaded K-contiguous into transposed LDS tiles ktT[d][k], vtT[e][k].
// D is compile-time so all fragment-array indices are static (no movrel).
// ---------------------------------------------------------------------------
template <int D>
__global__ __launch_bounds__(256) void attn_ctx_kernel(
    const bf16_t* __restrict__ KV, int R, int ld, int H, int koff, int voff,
    float scale, bf16_t* __restrict__ ctx) {
  constexpr int CTN = (D + 15) / 16;
  constexpr int NT  = (CTN * CTN + 7) / 8;
  constexpr int DP  = CTN * 16;  // 64 or 80
  __shared__ bf16_t ktT[DP][36];
  __shared__ bf16_t vtT[DP][36];
  __shared__ float  stage[DP][DP + 1];
  const int b = blockIdx.x / H, h = blockIdx.x % H;
  const int kc = koff + h * D, vc = voff + h * D;
  const int tid = threadIdx.x, wave = tid >> 5, lane = tid & 31;
  floatx8 acc[NT] = {};
  const size_t rowbase = (size_t)b * R;
  for (int k0 = 0; k0 < R; k0 += 32) {
    for (int i = tid; i < DP * 32; i += 256) {
      const int dd = i >> 5, kk = i & 31;
      const int gr = k0 + kk;
      bf16_t kv = (bf16_t)0.0f, vv = (bf16_t)0.0f;
      if (gr < R && dd < D) {
        const bf16_t* p = KV + (rowbase + gr) * (size_t)ld;
        kv = p[kc + dd]; vv = p[vc + dd];
      }
      ktT[dd][kk] = kv; vtT[dd][kk] = vv;
    }
    __syncthreads();
#pragma unroll
    for (int ai = 0; ai < NT; ++ai) {
      const int t = wave + ai * 8;
      if (t < CTN * CTN) {
        const int rt = t / CTN, ct = t % CTN;
        bf16x16 a  = load_a_frag(lane, &ktT[rt * 16][0], 36);
        bf16x16 bb = load_b_frag(lane, &vtT[ct * 16][0], 36);
        acc[ai] = wmma_bf16(a, bb, acc[ai]);
      }
    }
    __syncthreads();
  }
#pragma unroll
  for (int ai = 0; ai < NT; ++ai) {
    const int t = wave + ai * 8;
    if (t < CTN * CTN) {
      const int rt = t / CTN, ct = t % CTN;
      store_c(lane, acc[ai], [&](int r, int cc, float v) {
        stage[rt * 16 + r][ct * 16 + cc] = v * scale;
      });
    }
  }
  __syncthreads();
  if (tid < D) {
    const int col = tid;
    float mx = -1e30f;
    for (int r = 0; r < D; ++r) mx = fmaxf(mx, stage[r][col]);
    float sum = 0.f;
    for (int r = 0; r < D; ++r) sum += __expf(stage[r][col] - mx);
    const float inv = 1.f / sum;
    bf16_t* cp = ctx + (size_t)blockIdx.x * D * D;
    for (int r = 0; r < D; ++r)
      cp[r * D + col] = (bf16_t)(__expf(stage[r][col] - mx) * inv);
  }
}

// ---------------------------------------------------------------------------
// Spatial out: merge[b,n, h*64+e] = Q1 @ ctx2 + Q2 @ ctx1   (heads=8, d=64)
// grid: x = ceil(300/32), y = B*8.  8 waves cover 32(n) x 64(e).
// ---------------------------------------------------------------------------
__global__ __launch_bounds__(256) void attn_out_spatial_kernel(
    const bf16_t* __restrict__ S1, const bf16_t* __restrict__ S2,
    const bf16_t* __restrict__ ctx1, const bf16_t* __restrict__ ctx2,
    bf16_t* __restrict__ merge) {
  __shared__ bf16_t q1t[32][36], q2t[32][36];
  __shared__ bf16_t c1tT[64][36], c2tT[64][36];
  const int b = blockIdx.y >> 3, h = blockIdx.y & 7;
  const int n0 = blockIdx.x * 32;
  const int tid = threadIdx.x, wave = tid >> 5, lane = tid & 31;
  const int wr = wave & 1, wc = wave >> 1;
  const size_t cbase = (size_t)blockIdx.y * 4096;
  floatx8 acc = {};
  for (int k0 = 0; k0 < 64; k0 += 32) {
    for (int i = tid; i < 32 * 32; i += 256) {
      const int r = i >> 5, kk = i & 31;
      const int n = n0 + r;
      bf16_t v1 = (bf16_t)0.0f, v2 = (bf16_t)0.0f;
      if (n < kN) {
        const size_t q = ((size_t)b * kN + n) * kC + h * 64 + k0 + kk;
        v1 = S1[q]; v2 = S2[q];
      }
      q1t[r][kk] = v1; q2t[r][kk] = v2;
    }
    for (int i = tid; i < 64 * 32; i += 256) {
      const int e = i >> 5, kk = i & 31;
      c1tT[e][kk] = ctx1[cbase + (size_t)(k0 + kk) * 64 + e];
      c2tT[e][kk] = ctx2[cbase + (size_t)(k0 + kk) * 64 + e];
    }
    __syncthreads();
    bf16x16 a1 = load_a_frag(lane, &q1t[wr * 16][0], 36);
    bf16x16 b2 = load_b_frag(lane, &c2tT[wc * 16][0], 36);
    acc = wmma_bf16(a1, b2, acc);
    bf16x16 a2 = load_a_frag(lane, &q2t[wr * 16][0], 36);
    bf16x16 b1 = load_b_frag(lane, &c1tT[wc * 16][0], 36);
    acc = wmma_bf16(a2, b1, acc);
    __syncthreads();
  }
  store_c(lane, acc, [&](int r, int cc, float v) {
    const int n = n0 + wr * 16 + r;
    const int e = wc * 16 + cc;
    if (n < kN) merge[((size_t)b * kN + n) * 1024 + h * 64 + e] = (bf16_t)v;
  });
}

// ---------------------------------------------------------------------------
// Channel out: merge[b, h*75+e, 512+ch] = Qc1 @ ctxc2 + Qc2 @ ctxc1
// heads=4, d=75 (padded 80). grid: x = 512/32, y = B*4.  32(ch) x 80(e) tile,
// 10 tiles over 8 waves; indices fully static (unrolled, guarded).
// ---------------------------------------------------------------------------
__global__ __launch_bounds__(256) void attn_out_channel_kernel(
    const bf16_t* __restrict__ Cc1, const bf16_t* __restrict__ Cc2,
    const bf16_t* __restrict__ ctx1, const bf16_t* __restrict__ ctx2,
    bf16_t* __restrict__ merge) {
  __shared__ bf16_t q1t[32][36], q2t[32][36];
  __shared__ bf16_t c1tT[80][36], c2tT[80][36];
  const int b = blockIdx.y >> 2, h = blockIdx.y & 3;
  const int ch0 = blockIdx.x * 32;
  const int tid = threadIdx.x, wave = tid >> 5, lane = tid & 31;
  const size_t cbase = (size_t)blockIdx.y * 5625;
  floatx8 acc[2] = {};
  for (int k0 = 0; k0 < 75; k0 += 32) {
    for (int i = tid; i < 32 * 32; i += 256) {
      const int r = i >> 5, kk = i & 31;
      const int dd = k0 + kk;
      bf16_t v1 = (bf16_t)0.0f, v2 = (bf16_t)0.0f;
      if (dd < 75) {
        const size_t q = ((size_t)b * kC + ch0 + r) * kN + h * 75 + dd;
        v1 = Cc1[q]; v2 = Cc2[q];
      }
      q1t[r][kk] = v1; q2t[r][kk] = v2;
    }
    for (int i = tid; i < 80 * 32; i += 256) {
      const int e = i >> 5, kk = i & 31;
      const int dd = k0 + kk;
      bf16_t v1 = (bf16_t)0.0f, v2 = (bf16_t)0.0f;
      if (dd < 75 && e < 75) {
        v1 = ctx1[cbase + (size_t)dd * 75 + e];
        v2 = ctx2[cbase + (size_t)dd * 75 + e];
      }
      c1tT[e][kk] = v1; c2tT[e][kk] = v2;
    }
    __syncthreads();
#pragma unroll
    for (int ti = 0; ti < 2; ++ti) {
      const int t = wave + ti * 8;
      if (t < 10) {
        const int rt = t / 5, ct = t % 5;
        bf16x16 a1 = load_a_frag(lane, &q1t[rt * 16][0], 36);
        bf16x16 b2 = load_b_frag(lane, &c2tT[ct * 16][0], 36);
        acc[ti] = wmma_bf16(a1, b2, acc[ti]);
        bf16x16 a2 = load_a_frag(lane, &q2t[rt * 16][0], 36);
        bf16x16 b1 = load_b_frag(lane, &c1tT[ct * 16][0], 36);
        acc[ti] = wmma_bf16(a2, b1, acc[ti]);
      }
    }
    __syncthreads();
  }
#pragma unroll
  for (int ti = 0; ti < 2; ++ti) {
    const int t = wave + ti * 8;
    if (t < 10) {
      const int rt = t / 5, ct = t % 5;
      store_c(lane, acc[ti], [&](int r, int cc, float v) {
        const int ch = ch0 + rt * 16 + r;
        const int e = ct * 16 + cc;
        if (e < 75)
          merge[((size_t)b * kN + h * 75 + e) * 1024 + 512 + ch] = (bf16_t)v;
      });
    }
  }
}

// ---------------------------------------------------------------------------
// Depthwise 3x3 SAME + bias + LeakyReLU(0.01).  E1 row-major f32 -> E2 bf16.
// ---------------------------------------------------------------------------
__global__ __launch_bounds__(256) void dwconv_kernel(
    const float* __restrict__ E1, const float* __restrict__ dww,
    const float* __restrict__ dwb, bf16_t* __restrict__ E2) {
  const size_t i = (size_t)blockIdx.x * 256 + threadIdx.x;
  if (i >= (size_t)kBCN) return;
  const int ch = (int)(i % kC);
  const size_t t = i / kC;
  const int n = (int)(t % kN);
  const int b = (int)(t / kN);
  const int hh = n / kW, ww = n % kW;
  float acc = dwb[ch];
#pragma unroll
  for (int kh = 0; kh < 3; ++kh) {
    const int y = hh + kh - 1;
    if (y < 0 || y >= kH) continue;
#pragma unroll
    for (int kw2 = 0; kw2 < 3; ++kw2) {
      const int x = ww + kw2 - 1;
      if (x < 0 || x >= kW) continue;
      acc += E1[((size_t)b * kN + y * kW + x) * kC + ch] * dww[ch * 9 + kh * 3 + kw2];
    }
  }
  acc = acc >= 0.f ? acc : 0.01f * acc;
  E2[i] = (bf16_t)acc;
}

// ---------------------------------------------------------------------------
// Per-channel batchnorm stats over (B, N): X is (B, C, N) channel-major f32.
// ---------------------------------------------------------------------------
__global__ __launch_bounds__(256) void bn_stats_kernel(const float* __restrict__ X,
                                                       float* __restrict__ mean,
                                                       float* __restrict__ inv) {
  __shared__ float r1[256], r2[256];
  const int co = blockIdx.x, tid = threadIdx.x;
  float s = 0.f, s2 = 0.f;
  for (int i = tid; i < kB * kN; i += 256) {
    const int b = i / kN, n = i % kN;
    const float v = X[((size_t)b * kC + co) * kN + n];
    s += v; s2 += v * v;
  }
  r1[tid] = s; r2[tid] = s2;
  __syncthreads();
  for (int off = 128; off; off >>= 1) {
    if (tid < off) { r1[tid] += r1[tid + off]; r2[tid] += r2[tid + off]; }
    __syncthreads();
  }
  if (tid == 0) {
    const float m = r1[0] / (float)(kB * kN);
    const float v = r2[0] / (float)(kB * kN) - m * m;
    mean[co] = m;
    inv[co] = rsqrtf(v + 1e-5f);
  }
}

// y = resid + bn(E3).  resid row-major (B*N, C); E3/y channel-major (B, C, N).
__global__ __launch_bounds__(256) void bn_apply_add_kernel(
    const float* __restrict__ resid, const float* __restrict__ E3,
    const float* __restrict__ m, const float* __restrict__ inv,
    const float* __restrict__ w, const float* __restrict__ bb,
    float* __restrict__ Y) {
  const size_t i = (size_t)blockIdx.x * 256 + threadIdx.x;
  if (i >= (size_t)kBCN) return;
  const int n = (int)(i % kN);
  const int co = (int)((i / kN) % kC);
  const int b = (int)(i / ((size_t)kN * kC));
  const float e = (E3[i] - m[co]) * inv[co] * w[co] + bb[co];
  Y[i] = resid[((size_t)b * kN + n) * kC + co] + e;
}

__global__ __launch_bounds__(256) void bn_final_kernel(
    const float* __restrict__ Y, const float* __restrict__ m,
    const float* __restrict__ inv, const float* __restrict__ w,
    const float* __restrict__ bb, float* __restrict__ out) {
  const size_t i = (size_t)blockIdx.x * 256 + threadIdx.x;
  if (i >= (size_t)kBCN) return;
  const int co = (int)((i / kN) % kC);
  out[i] = (Y[i] - m[co]) * inv[co] * w[co] + bb[co];
}

// ---------------------------------------------------------------------------
// Host orchestration
// ---------------------------------------------------------------------------
extern "C" void kernel_launch(void* const* d_in, const int* in_sizes, int n_in,
                              void* d_out, int out_size, void* d_ws, size_t ws_size,
                              hipStream_t stream) {
  (void)in_sizes; (void)n_in; (void)out_size; (void)ws_size;
  const float* x1     = (const float*)d_in[0];
  const float* x2     = (const float*)d_in[1];
  const float* cp1_w  = (const float*)d_in[2];
  const float* cp1_b  = (const float*)d_in[3];
  const float* ln1_w  = (const float*)d_in[4];
  const float* ln1_b  = (const float*)d_in[5];
  const float* cp2_w  = (const float*)d_in[6];
  const float* cp2_b  = (const float*)d_in[7];
  const float* ln2_w  = (const float*)d_in[8];
  const float* ln2_b  = (const float*)d_in[9];
  const float* skv1   = (const float*)d_in[10];
  const float* skv2   = (const float*)d_in[11];
  const float* ckv1   = (const float*)d_in[12];
  const float* ckv2   = (const float*)d_in[13];
  const float* res_w  = (const float*)d_in[14];
  const float* ce1_w  = (const float*)d_in[15];
  const float* ce1_b  = (const float*)d_in[16];
  const float* dw_w   = (const float*)d_in[17];
  const float* dw_b   = (const float*)d_in[18];
  const float* ce3_w  = (const float*)d_in[19];
  const float* ce3_b  = (const float*)d_in[20];
  const float* bn_e_w = (const float*)d_in[21];
  const float* bn_e_b = (const float*)d_in[22];
  const float* bn_o_w = (const float*)d_in[23];
  const float* bn_o_b = (const float*)d_in[24];

  char* ws = (char*)d_ws;
  size_t off = 0;
  auto take = [&](size_t bytes) -> char* {
    char* p = ws + off;
    off += (bytes + 255) & ~((size_t)255);
    return p;
  };

  // bf16 weights (persistent)
  bf16_t* w_cp1  = (bf16_t*)take(1024 * 512 * 2);
  bf16_t* w_cp2  = (bf16_t*)take(1024 * 512 * 2);
  bf16_t* w_skv1 = (bf16_t*)take(1024 * 512 * 2);
  bf16_t* w_skv2 = (bf16_t*)take(1024 * 512 * 2);
  bf16_t* w_ckv1 = (bf16_t*)take(600 * 300 * 2);
  bf16_t* w_ckv2 = (bf16_t*)take(600 * 300 * 2);
  bf16_t* w_res  = (bf16_t*)take(512 * 1024 * 2);
  bf16_t* w_ce1  = (bf16_t*)take(512 * 1024 * 2);
  bf16_t* w_ce3  = (bf16_t*)take(512 * 512 * 2);

  // persistent activations
  bf16_t* S1  = (bf16_t*)take((size_t)kRows * 512 * 2);
  bf16_t* S2  = (bf16_t*)take((size_t)kRows * 512 * 2);
  bf16_t* Cc1 = (bf16_t*)take((size_t)kB * kC * kN * 2);
  bf16_t* Cc2 = (bf16_t*)take((size_t)kB * kC * kN * 2);
  bf16_t* ctx_s1 = (bf16_t*)take((size_t)kB * 8 * 64 * 64 * 2);
  bf16_t* ctx_s2 = (bf16_t*)take((size_t)kB * 8 * 64 * 64 * 2);
  bf16_t* ctx_c1 = (bf16_t*)take((size_t)kB * 4 * 75 * 75 * 2);
  bf16_t* ctx_c2 = (bf16_t*)take((size_t)kB * 4 * 75 * 75 * 2);
  float* mean_e = (float*)take(512 * 4);
  float* inv_e  = (float*)take(512 * 4);
  float* mean_o = (float*)take(512 * 4);
  float* inv_o  = (float*)take(512 * 4);

  // big phased-overlay region (157.3 MB)
  char* RB = take(157286400ULL);
  bf16_t* T1   = (bf16_t*)(RB);                 // phase 1
  bf16_t* T2   = (bf16_t*)(RB + 19660800);
  bf16_t* Yb   = (bf16_t*)(RB + 39321600);      // (19200x1024) bf16
  bf16_t* KV1  = (bf16_t*)(RB);                 // phase 2 (T,Y dead)
  bf16_t* KV2  = (bf16_t*)(RB + 39321600);
  bf16_t* KVc1 = (bf16_t*)(RB + 78643200);
  bf16_t* KVc2 = (bf16_t*)(RB + 117964800);
  bf16_t* merge = (bf16_t*)(RB);                // phase 3 (KV1 dead)
  float*  resid = (float*)(RB + 39321600);      // (KV2 dead)
  float*  E1    = (float*)(RB + 78643200);      // (KVc1 dead)
  bf16_t* E2    = (bf16_t*)(RB + 117964800);    // (KVc2 dead)
  float*  E3    = (float*)(RB);                 // (merge dead)
  float*  Ybn   = (float*)(RB + 78643200);      // (E1 dead)

  const int TPB = 256;
  const int gBCN = (int)((kBCN + TPB - 1) / TPB);

  // 1. weight conversion
  f32_to_bf16_kernel<<<2048, TPB, 0, stream>>>(cp1_w, w_cp1, 524288);
  f32_to_bf16_kernel<<<2048, TPB, 0, stream>>>(cp2_w, w_cp2, 524288);
  f32_to_bf16_kernel<<<2048, TPB, 0, stream>>>(skv1, w_skv1, 524288);
  f32_to_bf16_kernel<<<2048, TPB, 0, stream>>>(skv2, w_skv2, 524288);
  f32_to_bf16_kernel<<<704, TPB, 0, stream>>>(ckv1, w_ckv1, 180000);
  f32_to_bf16_kernel<<<704, TPB, 0, stream>>>(ckv2, w_ckv2, 180000);
  f32_to_bf16_kernel<<<2048, TPB, 0, stream>>>(res_w, w_res, 524288);
  f32_to_bf16_kernel<<<2048, TPB, 0, stream>>>(ce1_w, w_ce1, 524288);
  f32_to_bf16_kernel<<<1024, TPB, 0, stream>>>(ce3_w, w_ce3, 262144);

  // 2. pack inputs
  pack_kernel<<<gBCN, TPB, 0, stream>>>(x1, T1);
  pack_kernel<<<gBCN, TPB, 0, stream>>>(x2, T2);

  // 3. projection GEMM (+bias) -> LN+ReLU+split, stream 1 then stream 2
  gemm_kernel<1><<<dim3(300, 16, 1), TPB, 0, stream>>>(
      T1, 512, 0, w_cp1, 512, 0, Yb, 1024, 0, nullptr, cp1_b, kRows, 1024, 512);
  ln_relu_split_kernel<<<kRows, TPB, 0, stream>>>(Yb, ln1_w, ln1_b, S1, Cc1);
  gemm_kernel<1><<<dim3(300, 16, 1), TPB, 0, stream>>>(
      T2, 512, 0, w_cp2, 512, 0, Yb, 1024, 0, nullptr, cp2_b, kRows, 1024, 512);
  ln_relu_split_kernel<<<kRows, TPB, 0, stream>>>(Yb, ln2_w, ln2_b, S2, Cc2);

  // 4. KV projections (spatial: K=512; channel: K=300, M flattened over B*C)
  gemm_kernel<1><<<dim3(300, 16, 1), TPB, 0, stream>>>(
      S1, 512, 0, w_skv1, 512, 0, KV1, 1024, 0, nullptr, nullptr, kRows, 1024, 512);
  gemm_kernel<1><<<dim3(300, 16, 1), TPB, 0, stream>>>(
      S2, 512, 0, w_skv2, 512, 0, KV2, 1024, 0, nullptr, nullptr, kRows, 1024, 512);
  gemm_kernel<1><<<dim3(512, 10, 1), TPB, 0, stream>>>(
      Cc1, 300, 0, w_ckv1, 300, 0, KVc1, 600, 0, nullptr, nullptr, kB * kC, 600, 300);
  gemm_kernel<1><<<dim3(512, 10, 1), TPB, 0, stream>>>(
      Cc2, 300, 0, w_ckv2, 300, 0, KVc2, 600, 0, nullptr, nullptr, kB * kC, 600, 300);

  // 5. context matrices + softmax
  attn_ctx_kernel<64><<<kB * 8, TPB, 0, stream>>>(KV1, 300, 1024, 8, 0, 512,
                                                  0.125f, ctx_s1);
  attn_ctx_kernel<64><<<kB * 8, TPB, 0, stream>>>(KV2, 300, 1024, 8, 0, 512,
                                                  0.125f, ctx_s2);
  attn_ctx_kernel<75><<<kB * 4, TPB, 0, stream>>>(KVc1, 512, 600, 4, 0, 300,
                                                  0.11547005f, ctx_c1);
  attn_ctx_kernel<75><<<kB * 4, TPB, 0, stream>>>(KVc2, 512, 600, 4, 0, 300,
                                                  0.11547005f, ctx_c2);

  // 6. cross-direction outputs assemble merge (B*N, 1024)
  attn_out_spatial_kernel<<<dim3(10, kB * 8), TPB, 0, stream>>>(
      S1, S2, ctx_s1, ctx_s2, merge);
  attn_out_channel_kernel<<<dim3(16, kB * 4), TPB, 0, stream>>>(
      Cc1, Cc2, ctx_c1, ctx_c2, merge);

  // 7. residual + excitation path
  gemm_kernel<0><<<dim3(300, 8, 1), TPB, 0, stream>>>(
      merge, 1024, 0, w_res, 1024, 0, resid, 512, 0, nullptr, nullptr,
      kRows, 512, 1024);
  gemm_kernel<0><<<dim3(300, 8, 1), TPB, 0, stream>>>(
      merge, 1024, 0, w_ce1, 1024, 0, E1, 512, 0, nullptr, ce1_b,
      kRows, 512, 1024);
  dwconv_kernel<<<gBCN, TPB, 0, stream>>>(E1, dw_w, dw_b, E2);
  gemm_kernel<0><<<dim3(8, 5, 64), TPB, 0, stream>>>(
      w_ce3, 512, 0, E2, 512, (long long)kN * kC, E3, 300, (long long)kC * kN,
      ce3_b, nullptr, 512, 300, 512);

  // 8. batchnorms
  bn_stats_kernel<<<512, TPB, 0, stream>>>(E3, mean_e, inv_e);
  bn_apply_add_kernel<<<gBCN, TPB, 0, stream>>>(resid, E3, mean_e, inv_e,
                                                bn_e_w, bn_e_b, Ybn);
  bn_stats_kernel<<<512, TPB, 0, stream>>>(Ybn, mean_o, inv_o);
  bn_final_kernel<<<gBCN, TPB, 0, stream>>>(Ybn, mean_o, inv_o, bn_o_w, bn_o_b,
                                            (float*)d_out);
}
